// Self_Attention_69415261438649
// MI455X (gfx1250) — matrix operand order repaired and checked by reading, not compiled, not run on hardware
//
#include <hip/hip_runtime.h>

// Self-attention B=8, C=512, L=2048 on MI455X (gfx1250, wave32, WMMA f16->f32).
// All WMMA operands are laid out so each lane's fragment is a contiguous
// 16/32-byte load (A: rows K-contiguous; B: stored transposed [N][K]).
// Each GEMM wave computes a 32x64 register tile (2 M x 4 N accumulators),
// sharing B fragments across the two M rows: 8 WMMA per 12 b128 loads.

constexpr int kB  = 8;
constexpr int kC  = 512;
constexpr int kL  = 2048;
constexpr int kC8 = 64;

typedef __attribute__((ext_vector_type(16))) _Float16 v16h;
typedef __attribute__((ext_vector_type(8)))  _Float16 v8h;
typedef __attribute__((ext_vector_type(4)))  _Float16 v4h;
typedef __attribute__((ext_vector_type(8)))  float    v8f;
typedef __attribute__((ext_vector_type(4)))  float    v4f;

// ---------------------------------------------------------------------------
// WMMA fragment loads (wave32, 16x16x32 f16), A/B in this code's contiguous K order (same for both, not the ISA table order):
//  A (16x32): lane&15 = M; lane-half selects K 0-15 / 16-31, packed in order;
//             -> one contiguous 32B load from a K-contiguous row.
//  B (32x16): lane&15 = N; lane-half selects K 0-15 / 16-31, packed in order;
//             -> one contiguous 32B load from a [N][K] (transposed) row.
//  C/D f32:   lane&15 = N; VGPR r -> M = r + 8*(lane>=16).
// ---------------------------------------------------------------------------
__device__ inline v16h load_frag_A(const _Float16* __restrict__ rowk0, int lane) {
  int kh = (lane >> 4) * 16;
  return *(const v16h*)(rowk0 + kh);
}
__device__ inline v16h load_frag_B(const _Float16* __restrict__ rowk0, int lane) {
  int kh = (lane >> 4) * 16;
  return *(const v16h*)(rowk0 + kh);   // 32B contiguous
}
__device__ inline v8f wmma_f16(v16h a, v16h b, v8f c) {
  return __builtin_amdgcn_wmma_f32_16x16x32_f16(false, a, false, b,
                                                (short)0, c, false, false);
}

// ---------------------------------------------------------------------------
// Kernel 0a: elementwise f32 -> f16 convert (weights). n multiple of 1024.
// ---------------------------------------------------------------------------
__global__ __launch_bounds__(256) void convert_f16_kernel(
    const float* __restrict__ src, _Float16* __restrict__ dst) {
  int i = (blockIdx.x * 256 + threadIdx.x) * 4;
  v4f v = *(const v4f*)(src + i);
  v4h h;
#pragma unroll
  for (int q = 0; q < 4; ++q) h[q] = (_Float16)v[q];
  *(v4h*)(dst + i) = h;
}

// ---------------------------------------------------------------------------
// Kernel 0b: x [B][C][L] f32 -> xT [B][L][C] f16, 64x64 LDS-tiled transpose.
// ---------------------------------------------------------------------------
__global__ __launch_bounds__(256) void transpose_x_kernel(
    const float* __restrict__ x, _Float16* __restrict__ xT) {
  __shared__ float tile[64][65];
  int bat = blockIdx.x >> 8;          // / (8*32)
  int rem = blockIdx.x & 255;
  int c0 = (rem >> 5) * 64;           // over C (8 tiles)
  int l0 = (rem & 31) * 64;           // over L (32 tiles)
  int row = threadIdx.x >> 4;
  int col = (threadIdx.x & 15) * 4;
  const float* xb = x + (size_t)bat * kC * kL;
#pragma unroll
  for (int p = 0; p < 4; ++p) {
    const float* src = xb + (size_t)(c0 + row + p * 16) * kL + l0 + col;
#pragma unroll
    for (int q = 0; q < 4; ++q) tile[row + p * 16][col + q] = src[q];
  }
  __syncthreads();
  _Float16* xTb = xT + (size_t)bat * kL * kC;
#pragma unroll
  for (int p = 0; p < 4; ++p) {
    _Float16* dst = xTb + (size_t)(l0 + row + p * 16) * kC + c0 + col;
#pragma unroll
    for (int q = 0; q < 4; ++q)
      dst[q] = (_Float16)tile[col + q][row + p * 16];
  }
}

// ---------------------------------------------------------------------------
// Kernel 1: projection Y = Wh (COUT x C) * x_b + bias, operands f16.
// One wave computes a 32x64 strip (2 M x 4 N accumulators, shared B frags).
// TOUT: y stored [L][COUT] (q,k -> direct GEMM operands) else [COUT][L] (v).
// ---------------------------------------------------------------------------
template <int COUT, bool TOUT>
__global__ __launch_bounds__(256) void proj_kernel(
    const _Float16* __restrict__ Wh, const float* __restrict__ bias,
    const _Float16* __restrict__ xT, _Float16* __restrict__ y) {
  int lane = threadIdx.x & 31;
  int w = blockIdx.x * 8 + (threadIdx.x >> 5);
  constexpr int pairsM = COUT / 32;
  constexpr int quadsN = kL / 64;     // 32
  int wpb = pairsM * quadsN;
  int bat = w / wpb;
  int r = w % wpb;
  int m0 = (r / quadsN) * 32;
  int n0 = (r % quadsN) * 64;
  int n = lane & 15;

  const _Float16* xTb = xT + (size_t)bat * kL * kC;
  const _Float16* arow0 = Wh + (size_t)(m0 + n) * kC;        // lane&15 = M
  const _Float16* arow1 = arow0 + (size_t)16 * kC;
  v8f acc[2][4] = {};
  for (int k0 = 0; k0 < kC; k0 += 32) {
    v16h a0 = load_frag_A(arow0 + k0, lane);
    v16h a1 = load_frag_A(arow1 + k0, lane);
#pragma unroll
    for (int tq = 0; tq < 4; ++tq) {
      v16h b = load_frag_B(xTb + (size_t)(n0 + tq * 16 + n) * kC + k0, lane);
      acc[0][tq] = wmma_f16(a0, b, acc[0][tq]);
      acc[1][tq] = wmma_f16(a1, b, acc[1][tq]);
    }
  }
  int mb = (lane >> 4) * 8;
#pragma unroll
  for (int mi = 0; mi < 2; ++mi)
#pragma unroll
    for (int tq = 0; tq < 4; ++tq)
#pragma unroll
      for (int i = 0; i < 8; ++i) {
        int m = m0 + mi * 16 + mb + i;
        int nn = n0 + tq * 16 + n;
        float val = acc[mi][tq][i] + bias[m];
        if (TOUT)
          y[((size_t)bat * kL + nn) * COUT + m] = (_Float16)val;
        else
          y[((size_t)bat * COUT + m) * kL + nn] = (_Float16)val;
      }
}

// ---------------------------------------------------------------------------
// Kernel 2: S[i][j] = sum_c qT[i][c] * kT[j][c]   (K = 64 -> 2 WMMA steps)
// Wave computes 32(i) x 64(j).
// ---------------------------------------------------------------------------
__global__ __launch_bounds__(256) void scores_kernel(
    const _Float16* __restrict__ qT, const _Float16* __restrict__ kT,
    float* __restrict__ S) {
  int lane = threadIdx.x & 31;
  int w = blockIdx.x * 8 + (threadIdx.x >> 5);
  constexpr int wpb = (kL / 32) * (kL / 64);   // 2048
  int bat = w / wpb;
  int r = w % wpb;
  int i0 = (r / (kL / 64)) * 32;
  int j0 = (r % (kL / 64)) * 64;
  int n = lane & 15;

  const _Float16* qb = qT + (size_t)bat * kL * kC8;
  const _Float16* kb = kT + (size_t)bat * kL * kC8;
  const _Float16* arow0 = qb + (size_t)(i0 + n) * kC8;
  const _Float16* arow1 = arow0 + (size_t)16 * kC8;
  v8f acc[2][4] = {};
#pragma unroll
  for (int k0 = 0; k0 < kC8; k0 += 32) {
    v16h a0 = load_frag_A(arow0 + k0, lane);
    v16h a1 = load_frag_A(arow1 + k0, lane);
#pragma unroll
    for (int tq = 0; tq < 4; ++tq) {
      v16h b = load_frag_B(kb + (size_t)(j0 + tq * 16 + n) * kC8 + k0, lane);
      acc[0][tq] = wmma_f16(a0, b, acc[0][tq]);
      acc[1][tq] = wmma_f16(a1, b, acc[1][tq]);
    }
  }
  float* Sb = S + (size_t)bat * kL * kL;
  int mb = (lane >> 4) * 8;
#pragma unroll
  for (int mi = 0; mi < 2; ++mi)
#pragma unroll
    for (int tq = 0; tq < 4; ++tq)
#pragma unroll
      for (int i = 0; i < 8; ++i)
        Sb[(size_t)(i0 + mi * 16 + mb + i) * kL + j0 + tq * 16 + n] =
            acc[mi][tq][i];
}

// ---------------------------------------------------------------------------
// Kernel 3: in-place row softmax of S (S becomes A, f32). One block per row;
// exp is recomputed in the final pass (TRANS units) instead of stored,
// saving a 134MB round-trip.
// ---------------------------------------------------------------------------
__global__ __launch_bounds__(256) void softmax_kernel(float* __restrict__ S) {
  int bat = blockIdx.x >> 11;
  int row = blockIdx.x & (kL - 1);
  float* srow = S + ((size_t)bat * kL + row) * kL;
  __shared__ float red[256];
  int t = threadIdx.x;

  float m = -3.0e38f;
  for (int j = t; j < kL; j += 256) m = fmaxf(m, srow[j]);
  red[t] = m;
  __syncthreads();
  for (int s = 128; s > 0; s >>= 1) {
    if (t < s) red[t] = fmaxf(red[t], red[t + s]);
    __syncthreads();
  }
  float rowmax = red[0];
  __syncthreads();

  float sum = 0.0f;
  for (int j = t; j < kL; j += 256) sum += __expf(srow[j] - rowmax);
  red[t] = sum;
  __syncthreads();
  for (int s = 128; s > 0; s >>= 1) {
    if (t < s) red[t] += red[t + s];
    __syncthreads();
  }
  float inv = 1.0f / red[0];

  for (int j = t; j < kL; j += 256)
    srow[j] = __expf(srow[j] - rowmax) * inv;
}

// ---------------------------------------------------------------------------
// Kernel 4: 64x64 LDS-tiled transpose of A: writes attention_map = A^T (f32,
// final output) AND AhT = A^T (f16, the coalesced B-operand of GEMM3).
// ---------------------------------------------------------------------------
__global__ __launch_bounds__(256) void transpose_attn_kernel(
    const float* __restrict__ A, float* __restrict__ attnT,
    _Float16* __restrict__ AhT) {
  __shared__ float tile[64][65];
  int bat = blockIdx.x >> 10;          // / (32*32)
  int rem = blockIdx.x & 1023;
  int r0 = (rem >> 5) * 64;            // A row tile
  int c0 = (rem & 31) * 64;            // A col tile
  int row = threadIdx.x >> 4;
  int col = (threadIdx.x & 15) * 4;
  const float* Ab = A + (size_t)bat * kL * kL;
#pragma unroll
  for (int p = 0; p < 4; ++p) {
    const float* src = Ab + (size_t)(r0 + row + p * 16) * kL + c0 + col;
#pragma unroll
    for (int q = 0; q < 4; ++q) tile[row + p * 16][col + q] = src[q];
  }
  __syncthreads();
  float* tb = attnT + (size_t)bat * kL * kL;
  _Float16* hb = AhT + (size_t)bat * kL * kL;
#pragma unroll
  for (int p = 0; p < 4; ++p) {
    size_t o = (size_t)(c0 + row + p * 16) * kL + r0 + col;
#pragma unroll
    for (int q = 0; q < 4; ++q) {
      float v = tile[col + q][row + p * 16];
      tb[o + q] = v;
      hb[o + q] = (_Float16)v;
    }
  }
}

// ---------------------------------------------------------------------------
// Kernel 5: o[c][i] = sum_j v[c][j] * A[j][i] = sum_j vh[c][j] * AhT[i][j];
// out = x + gamma * o. Dominant GEMM: 64 K-steps x 8 WMMA per wave (32x64).
// ---------------------------------------------------------------------------
__global__ __launch_bounds__(256) void out_kernel(
    const _Float16* __restrict__ vh, const _Float16* __restrict__ AhT,
    const float* __restrict__ x, const float* __restrict__ gamma,
    float* __restrict__ out) {
  int lane = threadIdx.x & 31;
  int w = blockIdx.x * 8 + (threadIdx.x >> 5);
  constexpr int wpb = (kC / 32) * (kL / 64);   // 512
  int bat = w / wpb;
  int r = w % wpb;
  int m0 = (r / (kL / 64)) * 32;
  int n0 = (r % (kL / 64)) * 64;
  int n = lane & 15;

  const _Float16* vb = vh + (size_t)bat * kC * kL;     // [C][L]
  const _Float16* Ab = AhT + (size_t)bat * kL * kL;    // [i][j]
  const _Float16* arow0 = vb + (size_t)(m0 + n) * kL;
  const _Float16* arow1 = arow0 + (size_t)16 * kL;
  v8f acc[2][4] = {};
  for (int j0 = 0; j0 < kL; j0 += 32) {
    __builtin_prefetch(arow0 + j0 + 64, 0, 3);
    v16h a0 = load_frag_A(arow0 + j0, lane);
    v16h a1 = load_frag_A(arow1 + j0, lane);
#pragma unroll
    for (int tq = 0; tq < 4; ++tq) {
      v16h b = load_frag_B(Ab + (size_t)(n0 + tq * 16 + n) * kL + j0, lane);
      acc[0][tq] = wmma_f16(a0, b, acc[0][tq]);
      acc[1][tq] = wmma_f16(a1, b, acc[1][tq]);
    }
  }
  float g = gamma[0];
  int mb = (lane >> 4) * 8;
#pragma unroll
  for (int mi = 0; mi < 2; ++mi)
#pragma unroll
    for (int tq = 0; tq < 4; ++tq)
#pragma unroll
      for (int i = 0; i < 8; ++i) {
        size_t idx = ((size_t)bat * kC + (m0 + mi * 16 + mb + i)) * kL +
                     (n0 + tq * 16 + n);
        out[idx] = x[idx] + g * acc[mi][tq][i];
      }
}

// ---------------------------------------------------------------------------
extern "C" void kernel_launch(void* const* d_in, const int* in_sizes, int n_in,
                              void* d_out, int out_size, void* d_ws,
                              size_t ws_size, hipStream_t stream) {
  const float* x     = (const float*)d_in[0];
  const float* Wq    = (const float*)d_in[1];
  const float* bq    = (const float*)d_in[2];
  const float* Wk    = (const float*)d_in[3];
  const float* bk    = (const float*)d_in[4];
  const float* Wv    = (const float*)d_in[5];
  const float* bv    = (const float*)d_in[6];
  const float* gamma = (const float*)d_in[7];

  char* ws = (char*)d_ws;
  size_t off = 0;
  _Float16* xT  = (_Float16*)(ws + off); off += (size_t)kB * kL * kC * 2;   // 16 MB
  _Float16* Wqh = (_Float16*)(ws + off); off += (size_t)kC8 * kC * 2;       // 64 KB
  _Float16* Wkh = (_Float16*)(ws + off); off += (size_t)kC8 * kC * 2;       // 64 KB
  _Float16* Wvh = (_Float16*)(ws + off); off += (size_t)kC * kC * 2;        // 512 KB
  _Float16* qT  = (_Float16*)(ws + off); off += (size_t)kB * kL * kC8 * 2;  // 2 MB
  _Float16* kT  = (_Float16*)(ws + off); off += (size_t)kB * kL * kC8 * 2;  // 2 MB
  _Float16* vh  = (_Float16*)(ws + off); off += (size_t)kB * kC * kL * 2;   // 16 MB
  _Float16* AhT = (_Float16*)(ws + off); off += (size_t)kB * kL * kL * 2;   // 64 MB
  float*    S   = (float*)(ws + off);    off += (size_t)kB * kL * kL * 4;   // 128 MB

  float* outMain = (float*)d_out;                    // [B][C][L]
  float* attnT   = outMain + (size_t)kB * kC * kL;   // [B][L][L] = A^T

  // 0) f16 staging: weights + transposed activations
  convert_f16_kernel<<<(kC8 * kC) / 1024, 256, 0, stream>>>(Wq, Wqh);
  convert_f16_kernel<<<(kC8 * kC) / 1024, 256, 0, stream>>>(Wk, Wkh);
  convert_f16_kernel<<<(kC * kC) / 1024, 256, 0, stream>>>(Wv, Wvh);
  transpose_x_kernel<<<kB * (kC / 64) * (kL / 64), 256, 0, stream>>>(x, xT);

  // 1) projections: q,k -> [L][64] (GEMM operand layout), v -> [C][L]
  proj_kernel<kC8, true><<<(kB * (kC8 / 32) * (kL / 64)) / 8, 256, 0, stream>>>(
      Wqh, bq, xT, qT);
  proj_kernel<kC8, true><<<(kB * (kC8 / 32) * (kL / 64)) / 8, 256, 0, stream>>>(
      Wkh, bk, xT, kT);
  proj_kernel<kC, false><<<(kB * (kC / 32) * (kL / 64)) / 8, 256, 0, stream>>>(
      Wvh, bv, xT, vh);

  // 2) scores, 3) softmax (in place), 4) transpose -> attn output + f16 operand
  scores_kernel<<<(kB * (kL / 32) * (kL / 64)) / 8, 256, 0, stream>>>(qT, kT, S);
  softmax_kernel<<<kB * kL, 256, 0, stream>>>(S);
  transpose_attn_kernel<<<kB * (kL / 64) * (kL / 64), 256, 0, stream>>>(
      S, attnT, AhT);

  // 5) output GEMM + residual
  out_kernel<<<(kB * (kC / 32) * (kL / 64)) / 8, 256, 0, stream>>>(
      vh, AhT, x, gamma, outMain);
}
